// GNNLight_42442866819689
// MI455X (gfx1250) — matrix operand order repaired and checked by reading, not compiled, not run on hardware
//
#include <hip/hip_runtime.h>
#include <hip/hip_bf16.h>

typedef __attribute__((ext_vector_type(16))) _Float16 v16h;
typedef __attribute__((ext_vector_type(8)))  _Float16 v8h;
typedef __attribute__((ext_vector_type(4)))  _Float16 v4h;
typedef __attribute__((ext_vector_type(8)))  float    v8f;
typedef __attribute__((ext_vector_type(4)))  float    v4f;

#define BSZ 2
#define NN  50000
#define NED 400000
#define DD  128
#define DEE 128
#define HH  64
#define HAT 16

__device__ __forceinline__ float silu_f(float x) { return x / (1.0f + __expf(-x)); }

__device__ __forceinline__ v8f wmma_f16(v16h a, v16h b, v8f c) {
    return __builtin_amdgcn_wmma_f32_16x16x32_f16(false, a, false, b, (short)0, c, false, false);
}

// A-fragment (16x32 f16, MxK) from LDS row-major tile.  Lane L holds row M=L%16;
// halves 0..7 = K[k0 + (L/16)*8 .. +7], halves 8..15 = K[k0+16+(L/16)*8 .. +7].
__device__ __forceinline__ v16h load_a_lds(const _Float16* base, int stride, int lane, int k0) {
    const _Float16* p = base + (lane & 15) * stride + k0 + (lane >> 4) * 8;
    v8h lo = *(const v8h*)p;
    v8h hi = *(const v8h*)(p + 16);
    v16h a;
#pragma unroll
    for (int i = 0; i < 8; ++i) { a[i] = lo[i]; a[i + 8] = hi[i]; }
    return a;
}

// B-fragment (32x16 f16, KxN) from global transposed weight WT[N][K].
// Lane L holds column N=n0+L%16; halves 0..7 = K[k0 + (L/16)*8 .. +7], halves 8..15 = K[k0+16+(L/16)*8 .. +7].
__device__ __forceinline__ v16h load_b_wt(const _Float16* wt, int K, int lane, int n0, int k0) {
    const _Float16* p = wt + (size_t)(n0 + (lane & 15)) * K + k0 + (lane >> 4) * 8;
    v8h c0 = *(const v8h*)p;
    v8h c1 = *(const v8h*)(p + 16);
    v16h b;
#pragma unroll
    for (int i = 0; i < 8; ++i) { b[i] = c0[i]; b[i + 8] = c1[i]; }
    return b;
}

__device__ __forceinline__ void store4h(_Float16* p, v4f x) {
    v4h h;
    h.x = (_Float16)x.x; h.y = (_Float16)x.y; h.z = (_Float16)x.z; h.w = (_Float16)x.w;
    *(v4h*)p = h;
}

__device__ __forceinline__ void atomic_add_f32(float* p, float v) {
    __hip_atomic_fetch_add(p, v, __ATOMIC_RELAXED, __HIP_MEMORY_SCOPE_AGENT);
}

// ---------------- prep: transpose f32 weights [K][Nw] -> f16 WT [Nw][K] ----------------
__global__ void wt_transpose_kernel(const float* __restrict__ src, _Float16* __restrict__ dst,
                                    int K, int Nw) {
    int i = blockIdx.x * blockDim.x + threadIdx.x;
    if (i < K * Nw) {
        int k = i / Nw, n = i % Nw;
        dst[(size_t)n * K + k] = (_Float16)src[i];
    }
}

__global__ void zero_kernel(float* __restrict__ p, size_t n) {
    size_t i = (size_t)blockIdx.x * blockDim.x + threadIdx.x;
    if (i < n) p[i] = 0.0f;
}

// ---------------- edge pass: msg MLP, attention exp, atomics, edge_delta ----------------
__launch_bounds__(64)
__global__ void edge_kernel(const float* __restrict__ V, const float* __restrict__ E,
                            const int* __restrict__ edges,
                            const _Float16* __restrict__ Wm1T, const float* __restrict__ bm1,
                            const _Float16* __restrict__ Wm2T, const float* __restrict__ bm2,
                            const _Float16* __restrict__ Wa1T, const float* __restrict__ ba1,
                            const float* __restrict__ Wa2, const float* __restrict__ ba2,
                            const _Float16* __restrict__ Wu1T, const float* __restrict__ bu1,
                            const _Float16* __restrict__ Wu2T, const float* __restrict__ bu2,
                            float* __restrict__ den, float* __restrict__ agg,
                            float* __restrict__ delta_out) {
    __shared__ _Float16 s_in[2][16 * 392];    // edge_inpt tile (f16), padded stride
    __shared__ _Float16 s_h1[2][16 * 72];     // hidden(64) tile
    __shared__ _Float16 s_mh[2][16 * 136];    // edge_msg f16 (A-frag source for attn)
    __shared__ float    s_mf[2][16 * 132];    // edge_msg f32 (for aggregation atomics)
    __shared__ float    s_ha[2][16 * 17];     // attn hidden(16)
    __shared__ float    s_ex[2][16];
    __shared__ int      s_rc[2][16];

    const int wave = threadIdx.x >> 5;
    const int lane = threadIdx.x & 31;
    const int tile = blockIdx.x * 2 + wave;
    const int tpb  = NED / 16;
    const int b    = tile / tpb;
    const int e0   = (tile % tpb) * 16;

    _Float16* in = s_in[wave];
    _Float16* h1 = s_h1[wave];
    _Float16* mh = s_mh[wave];
    float*    mf = s_mf[wave];

    // ---- stage edge_inpt = [sender | receiver | E] as f16 into LDS ----
    for (int m = 0; m < 16; ++m) {
        size_t eidx = (size_t)b * NED + e0 + m;
        int s = edges[eidx * 2 + 0];
        int r = edges[eidx * 2 + 1];
        v4f x;
        x = *(const v4f*)(V + ((size_t)b * NN + s) * DD + lane * 4);
        store4h(in + m * 392 + lane * 4, x);
        x = *(const v4f*)(V + ((size_t)b * NN + r) * DD + lane * 4);
        store4h(in + m * 392 + 128 + lane * 4, x);
        x = *(const v4f*)(E + eidx * DEE + lane * 4);
        store4h(in + m * 392 + 256 + lane * 4, x);
    }
    if (lane < 16) s_rc[wave][lane] = edges[((size_t)b * NED + e0 + lane) * 2 + 1];
    __builtin_prefetch(edges + ((size_t)b * NED + e0 + 16) * 2, 0, 1);
    __syncthreads();

    const int n  = lane & 15;
    const int hl = lane >> 4;

    // ---- GEMM1: [16x384] @ [384x64], SiLU ----
    v8f acc[4];
#pragma unroll
    for (int t = 0; t < 4; ++t) acc[t] = (v8f){};
    for (int kb = 0; kb < 12; ++kb) {
        v16h a = load_a_lds(in, 392, lane, kb * 32);
#pragma unroll
        for (int t = 0; t < 4; ++t)
            acc[t] = wmma_f16(a, load_b_wt(Wm1T, 384, lane, t * 16, kb * 32), acc[t]);
    }
#pragma unroll
    for (int t = 0; t < 4; ++t) {
        float bias = bm1[t * 16 + n];
#pragma unroll
        for (int v = 0; v < 8; ++v)
            h1[(v + 8 * hl) * 72 + t * 16 + n] = (_Float16)silu_f(acc[t][v] + bias);
    }
    __syncthreads();

    // ---- GEMM2: [16x64] @ [64x128] -> edge_msg ----
    v8f acc2[8];
#pragma unroll
    for (int t = 0; t < 8; ++t) acc2[t] = (v8f){};
    for (int kb = 0; kb < 2; ++kb) {
        v16h a = load_a_lds(h1, 72, lane, kb * 32);
#pragma unroll
        for (int t = 0; t < 8; ++t)
            acc2[t] = wmma_f16(a, load_b_wt(Wm2T, 64, lane, t * 16, kb * 32), acc2[t]);
    }
#pragma unroll
    for (int t = 0; t < 8; ++t) {
        float bias = bm2[t * 16 + n];
#pragma unroll
        for (int v = 0; v < 8; ++v) {
            float x = acc2[t][v] + bias;
            int m = v + 8 * hl;
            mh[m * 136 + t * 16 + n] = (_Float16)x;
            mf[m * 132 + t * 16 + n] = x;
        }
    }
    __syncthreads();

    // ---- attention: silu(msg @ Wa1 + ba1) @ Wa2 + ba2, clip, exp ----
    v8f accA = (v8f){};
    for (int kb = 0; kb < 4; ++kb) {
        v16h a = load_a_lds(mh, 136, lane, kb * 32);
        accA = wmma_f16(a, load_b_wt(Wa1T, 128, lane, 0, kb * 32), accA);
    }
#pragma unroll
    for (int v = 0; v < 8; ++v)
        s_ha[wave][(v + 8 * hl) * 17 + n] = silu_f(accA[v] + ba1[n]);
    __syncthreads();
    if (lane < 16) {
        float lg = ba2[0];
#pragma unroll
        for (int j = 0; j < 16; ++j) lg += s_ha[wave][lane * 17 + j] * Wa2[j];
        lg = fminf(30.0f, fmaxf(-30.0f, lg));
        float ex = __expf(lg);
        s_ex[wave][lane] = ex;
        atomic_add_f32(den + (size_t)b * NN + s_rc[wave][lane], ex);
    }
    __syncthreads();

    // ---- numerator: agg[recv] += ex * msg (hardware f32 atomics) ----
    for (int m = 0; m < 16; ++m) {
        float ex = s_ex[wave][m];
        int r = s_rc[wave][m];
        float* dst = agg + ((size_t)b * NN + r) * DD + lane * 4;
        const float* src = mf + m * 132 + lane * 4;
#pragma unroll
        for (int j = 0; j < 4; ++j) atomic_add_f32(dst + j, ex * src[j]);
    }
    __syncthreads();

    // ---- edge_delta: same edge_inpt through update MLP ----
    v8f accd[4];
#pragma unroll
    for (int t = 0; t < 4; ++t) accd[t] = (v8f){};
    for (int kb = 0; kb < 12; ++kb) {
        v16h a = load_a_lds(in, 392, lane, kb * 32);
#pragma unroll
        for (int t = 0; t < 4; ++t)
            accd[t] = wmma_f16(a, load_b_wt(Wu1T, 384, lane, t * 16, kb * 32), accd[t]);
    }
#pragma unroll
    for (int t = 0; t < 4; ++t) {
        float bias = bu1[t * 16 + n];
#pragma unroll
        for (int v = 0; v < 8; ++v)
            h1[(v + 8 * hl) * 72 + t * 16 + n] = (_Float16)silu_f(accd[t][v] + bias);
    }
    __syncthreads();

    v8f accd2[8];
#pragma unroll
    for (int t = 0; t < 8; ++t) accd2[t] = (v8f){};
    for (int kb = 0; kb < 2; ++kb) {
        v16h a = load_a_lds(h1, 72, lane, kb * 32);
#pragma unroll
        for (int t = 0; t < 8; ++t)
            accd2[t] = wmma_f16(a, load_b_wt(Wu2T, 64, lane, t * 16, kb * 32), accd2[t]);
    }
#pragma unroll
    for (int t = 0; t < 8; ++t) {
        float bias = bu2[t * 16 + n];
#pragma unroll
        for (int v = 0; v < 8; ++v) {
            int m = v + 8 * hl;
            delta_out[((size_t)b * NED + e0 + m) * DD + t * 16 + n] = accd2[t][v] + bias;
        }
    }
}

// ---------------- node pass: normalize aggregation, node MLP ----------------
__launch_bounds__(64)
__global__ void node_kernel(const float* __restrict__ V, const float* __restrict__ den,
                            const float* __restrict__ agg,
                            const _Float16* __restrict__ Wn1T, const float* __restrict__ bn1,
                            const _Float16* __restrict__ Wn2T, const float* __restrict__ bn2,
                            float* __restrict__ node_out) {
    __shared__ _Float16 s_in[2][16 * 264];
    __shared__ _Float16 s_h1[2][16 * 72];

    const int wave = threadIdx.x >> 5;
    const int lane = threadIdx.x & 31;
    const int tile = blockIdx.x * 2 + wave;
    const int tpb  = NN / 16;
    const int b    = tile / tpb;
    const int n0   = (tile % tpb) * 16;

    _Float16* in = s_in[wave];
    _Float16* h1 = s_h1[wave];

    for (int m = 0; m < 16; ++m) {
        size_t nid = (size_t)b * NN + n0 + m;
        v4f x = *(const v4f*)(V + nid * DD + lane * 4);
        store4h(in + m * 264 + lane * 4, x);
        float d = den[nid];
        float inv = (d > 0.0f) ? (1.0f / d) : 0.0f;
        v4f g = *(const v4f*)(agg + nid * DD + lane * 4);
        g *= inv;
        store4h(in + m * 264 + 128 + lane * 4, g);
    }
    __syncthreads();

    const int n  = lane & 15;
    const int hl = lane >> 4;

    v8f acc[4];
#pragma unroll
    for (int t = 0; t < 4; ++t) acc[t] = (v8f){};
    for (int kb = 0; kb < 8; ++kb) {
        v16h a = load_a_lds(in, 264, lane, kb * 32);
#pragma unroll
        for (int t = 0; t < 4; ++t)
            acc[t] = wmma_f16(a, load_b_wt(Wn1T, 256, lane, t * 16, kb * 32), acc[t]);
    }
#pragma unroll
    for (int t = 0; t < 4; ++t) {
        float bias = bn1[t * 16 + n];
#pragma unroll
        for (int v = 0; v < 8; ++v)
            h1[(v + 8 * hl) * 72 + t * 16 + n] = (_Float16)silu_f(acc[t][v] + bias);
    }
    __syncthreads();

    v8f acc2[8];
#pragma unroll
    for (int t = 0; t < 8; ++t) acc2[t] = (v8f){};
    for (int kb = 0; kb < 2; ++kb) {
        v16h a = load_a_lds(h1, 72, lane, kb * 32);
#pragma unroll
        for (int t = 0; t < 8; ++t)
            acc2[t] = wmma_f16(a, load_b_wt(Wn2T, 64, lane, t * 16, kb * 32), acc2[t]);
    }
#pragma unroll
    for (int t = 0; t < 8; ++t) {
        float bias = bn2[t * 16 + n];
#pragma unroll
        for (int v = 0; v < 8; ++v) {
            int m = v + 8 * hl;
            node_out[((size_t)b * NN + n0 + m) * DD + t * 16 + n] = acc2[t][v] + bias;
        }
    }
}

extern "C" void kernel_launch(void* const* d_in, const int* in_sizes, int n_in,
                              void* d_out, int out_size, void* d_ws, size_t ws_size,
                              hipStream_t stream) {
    const float* V     = (const float*)d_in[0];
    const float* E     = (const float*)d_in[1];
    const int*   edges = (const int*)d_in[2];
    const float* Wm1 = (const float*)d_in[3];  const float* bm1 = (const float*)d_in[4];
    const float* Wm2 = (const float*)d_in[5];  const float* bm2 = (const float*)d_in[6];
    const float* Wa1 = (const float*)d_in[7];  const float* ba1 = (const float*)d_in[8];
    const float* Wa2 = (const float*)d_in[9];  const float* ba2 = (const float*)d_in[10];
    const float* Wu1 = (const float*)d_in[11]; const float* bu1 = (const float*)d_in[12];
    const float* Wu2 = (const float*)d_in[13]; const float* bu2 = (const float*)d_in[14];
    const float* Wn1 = (const float*)d_in[15]; const float* bn1 = (const float*)d_in[16];
    const float* Wn2 = (const float*)d_in[17]; const float* bn2 = (const float*)d_in[18];

    char* ws = (char*)d_ws;
    size_t off = 0;
    _Float16* Wm1T = (_Float16*)(ws + off); off += (size_t)64 * 384 * 2;   // 49152
    _Float16* Wm2T = (_Float16*)(ws + off); off += (size_t)128 * 64 * 2;   // 16384
    _Float16* Wa1T = (_Float16*)(ws + off); off += (size_t)16 * 128 * 2;   // 4096
    _Float16* Wu1T = (_Float16*)(ws + off); off += (size_t)64 * 384 * 2;
    _Float16* Wu2T = (_Float16*)(ws + off); off += (size_t)128 * 64 * 2;
    _Float16* Wn1T = (_Float16*)(ws + off); off += (size_t)64 * 256 * 2;
    _Float16* Wn2T = (_Float16*)(ws + off); off += (size_t)128 * 64 * 2;
    float* den = (float*)(ws + off); off += (size_t)BSZ * NN * 4;
    float* agg = (float*)(ws + off); off += (size_t)BSZ * NN * DD * 4;

    // weight transposes (f32 -> f16 [N][K]); tiny, L2-resident afterwards
    auto tlaunch = [&](const float* s, _Float16* d, int K, int Nw) {
        int tot = K * Nw;
        wt_transpose_kernel<<<(tot + 255) / 256, 256, 0, stream>>>(s, d, K, Nw);
    };
    tlaunch(Wm1, Wm1T, 384, 64);
    tlaunch(Wm2, Wm2T, 64, 128);
    tlaunch(Wa1, Wa1T, 128, 16);
    tlaunch(Wu1, Wu1T, 384, 64);
    tlaunch(Wu2, Wu2T, 64, 128);
    tlaunch(Wn1, Wn1T, 256, 64);
    tlaunch(Wn2, Wn2T, 64, 128);

    // zero den+agg (contiguous)
    size_t nz = (size_t)BSZ * NN * (1 + DD);
    zero_kernel<<<(unsigned)((nz + 255) / 256), 256, 0, stream>>>(den, nz);

    float* node_out  = (float*)d_out;
    float* delta_out = node_out + (size_t)BSZ * NN * DD;

    // 2 waves/block, 16 rows per wave; grids divide exactly (400000*2/32, 50000*2/32)
    edge_kernel<<<(BSZ * NED) / 32, 64, 0, stream>>>(
        V, E, edges, Wm1T, bm1, Wm2T, bm2, Wa1T, ba1, Wa2, ba2,
        Wu1T, bu1, Wu2T, bu2, den, agg, delta_out);

    node_kernel<<<(BSZ * NN) / 32, 64, 0, stream>>>(
        V, den, agg, Wn1T, bn1, Wn2T, bn2, node_out);
}
